// RandomWanTransformer_89773406421156
// MI455X (gfx1250) — matrix operand order repaired and checked by reading, not compiled, not run on hardware
//
#include <hip/hip_runtime.h>
#include <cstdint>
#include <math.h>

// ---------------- constants (from the reference) ----------------
#define DMODEL 1536
#define NHEAD  12
#define DHEAD  128
#define NLAYER 2
#define FFDIM  6144
#define TDIM   4096
#define TLEN   512
#define FDIM   256
#define SEQ    896      // NF * (HI/2) * (WI/2) = 2*16*28
#define CURTOK 896      // current_start
#define KVTOT  1792     // current_start + SEQ
#define KVCAP  2688
#define COUT_C 16
#define NF_F   2
#define HI_H   32
#define WI_W   56

typedef __attribute__((ext_vector_type(16))) __bf16 v16bf;
typedef __attribute__((ext_vector_type(8)))  float  v8f;

union FragBF { v16bf v; uint32_t u[8]; };

// native converts -> v_cvt_pk_bf16_f32 style codegen instead of manual rounding
static __device__ __forceinline__ uint32_t bfpack(float2 f) {
    union { __bf16 h[2]; uint32_t u; } c;
    c.h[0] = (__bf16)f.x; c.h[1] = (__bf16)f.y;
    return c.u;
}
static __device__ __forceinline__ uint16_t bfbits(float f) {
    union { __bf16 b; uint16_t u; } c; c.b = (__bf16)f; return c.u;
}

static __device__ __forceinline__ float gelu_tanh(float x) {
    const float k0 = 0.7978845608028654f, k1 = 0.044715f;
    float x3 = x * x * x;
    return 0.5f * x * (1.0f + tanhf(k0 * (x + k1 * x3)));
}

// =================================================================
// Core: batched GEMM, f32 in / f32 out, bf16 WMMA compute.
// Tile: 128(M) x BN(N), K-step 32, 256 threads = 8 wave32 waves.
// Wave w owns rows [w*16, w*16+16); BN/16 accumulators cover BN cols.
// Software pipeline: float2 register staging + LDS ping-pong so the
// next K-slab's global loads overlap the current slab's WMMAs.
// Epilogue: +bias[n], optional GELU, *scaleVec[n], +residual.
// Requires M%128==0, N%BN==0, K%32==0, 8B-aligned A/B (true here).
// =================================================================
template <int BN>
__global__ __launch_bounds__(256) void gemm_wmma_bf16(
    const float* __restrict__ A, long lda, long strideA,
    const float* __restrict__ B, long ldb, long strideB, int transB,
    float* __restrict__ C, long ldc, long strideC,
    const float* __restrict__ bias,
    const float* __restrict__ scaleVec,
    const float* __restrict__ residual,
    int M, int N, int K, int act)
{
    constexpr int NFRAG = BN / 16;   // accumulators per wave
    constexpr int BPT   = BN / 16;   // B float2-pairs per thread per slab
    __shared__ uint16_t Als[2 * 128 * 34];   // bf16, padded stride 34 (17 dwords)
    __shared__ uint16_t Bls[2 * BN * 34];    // stored [n][k]: k-pairs contiguous

    const int tid  = threadIdx.x;
    const int lane = tid & 31;
    const int wave = tid >> 5;
    const int half = lane >> 4;
    const int l15  = lane & 15;

    const long zA = (long)blockIdx.z * strideA;
    const long zB = (long)blockIdx.z * strideB;
    const long zC = (long)blockIdx.z * strideC;
    const int m0t = blockIdx.y * 128;
    const int n0t = blockIdx.x * BN;

    float2 aST[8];     // A slab: 128x32 f32 = 2048 pairs / 256 thr
    float2 bST[BPT];   // B slab: 32xBN f32 = 16*BN pairs / 256 thr

    auto loadA = [&](int k0) {
        #pragma unroll
        for (int i = 0; i < 8; ++i) {
            int p = tid + i * 256;
            int m = p >> 4, kp = (p & 15) << 1;
            aST[i] = *(const float2*)&A[zA + (long)(m0t + m) * lda + (k0 + kp)];
        }
    };
    auto loadB = [&](int k0) {
        #pragma unroll
        for (int i = 0; i < BPT; ++i) {
            int p = tid + i * 256;
            if (transB) {
                int n = p >> 4, kp = (p & 15) << 1;
                bST[i] = *(const float2*)&B[zB + (long)(n0t + n) * ldb + (k0 + kp)];
            } else {
                int k = p / (BN / 2), np = (p % (BN / 2)) << 1;
                bST[i] = *(const float2*)&B[zB + (long)(k0 + k) * ldb + (n0t + np)];
            }
        }
    };
    auto storeA = [&](int buf) {
        uint16_t* dst = &Als[buf * 128 * 34];
        #pragma unroll
        for (int i = 0; i < 8; ++i) {
            int p = tid + i * 256;
            int m = p >> 4, kp = (p & 15) << 1;
            *(uint32_t*)&dst[m * 34 + kp] = bfpack(aST[i]);
        }
    };
    auto storeB = [&](int buf) {
        uint16_t* dst = &Bls[buf * BN * 34];
        #pragma unroll
        for (int i = 0; i < BPT; ++i) {
            int p = tid + i * 256;
            if (transB) {
                int n = p >> 4, kp = (p & 15) << 1;
                *(uint32_t*)&dst[n * 34 + kp] = bfpack(bST[i]);
            } else {
                int k = p / (BN / 2), np = (p % (BN / 2)) << 1;
                dst[np * 34 + k]       = bfbits(bST[i].x);
                dst[(np + 1) * 34 + k] = bfbits(bST[i].y);
            }
        }
    };

    v8f acc[NFRAG] = {};

    // prologue: stage slab 0
    loadA(0); loadB(0);
    storeA(0); storeB(0);

    int cur = 0;
    for (int k0 = 0; k0 < K; k0 += 32) {
        __syncthreads();
        const bool more = (k0 + 32) < K;
        if (more) { loadA(k0 + 32); loadB(k0 + 32); }   // overlap with WMMA below

        const uint16_t* Ab = &Als[cur * 128 * 34];
        const uint16_t* Bb = &Bls[cur * BN * 34];

        // A fragment: 16x32, rows = lane&15, K split per ISA 7.12.2
        FragBF a;
        const int arow = wave * 16 + l15;
        #pragma unroll
        for (int v = 0; v < 8; ++v) {
            int kk = ((v & 3) << 1) + ((v >> 2) << 4) + (half << 3);
            a.u[v] = *(const uint32_t*)&Ab[arow * 34 + kk];
        }
        #pragma unroll
        for (int f = 0; f < NFRAG; ++f) {
            FragBF b;
            const int bn = f * 16 + l15;
            #pragma unroll
            for (int v = 0; v < 8; ++v) {
                int kk = ((v & 3) << 1) + ((v >> 2) << 4) + (half << 3);
                b.u[v] = *(const uint32_t*)&Bb[bn * 34 + kk];
            }
            acc[f] = __builtin_amdgcn_wmma_f32_16x16x32_bf16(
                false, a.v, false, b.v, (short)0, acc[f], false, false);
        }

        if (more) { storeA(cur ^ 1); storeB(cur ^ 1); }
        cur ^= 1;
    }

    // ---- epilogue (C layout: vgpr r -> M=r or r+8; lane&15 -> N) ----
    #pragma unroll
    for (int f = 0; f < NFRAG; ++f) {
        const int gn = n0t + f * 16 + l15;
        #pragma unroll
        for (int r = 0; r < 8; ++r) {
            const int gm = m0t + wave * 16 + half * 8 + r;
            float val = acc[f][r];
            if (bias)     val += bias[gn];
            if (act == 1) val  = gelu_tanh(val);
            if (scaleVec) val *= scaleVec[gn];
            const long idx = zC + (long)gm * ldc + gn;
            if (residual) val += residual[idx];
            C[idx] = val;
        }
    }
}

// ---------------- row-wise LayerNorm * gamma(+1) + beta ----------------
__global__ __launch_bounds__(256) void layernorm_mod(
    const float* __restrict__ x, float* __restrict__ out,
    const float* __restrict__ gamma, const float* __restrict__ beta,
    int addOne, int N)
{
    const int row = blockIdx.x;
    const float* xr = x + (long)row * N;
    float* orw = out + (long)row * N;
    __shared__ float red[256];
    float s = 0.f, ss = 0.f;
    for (int i = threadIdx.x; i < N; i += 256) { float v = xr[i]; s += v; ss += v * v; }
    red[threadIdx.x] = s; __syncthreads();
    for (int o = 128; o > 0; o >>= 1) { if (threadIdx.x < o) red[threadIdx.x] += red[threadIdx.x + o]; __syncthreads(); }
    const float mu = red[0] / N; __syncthreads();
    red[threadIdx.x] = ss; __syncthreads();
    for (int o = 128; o > 0; o >>= 1) { if (threadIdx.x < o) red[threadIdx.x] += red[threadIdx.x + o]; __syncthreads(); }
    const float var = red[0] / N - mu * mu;
    const float inv = rsqrtf(var + 1e-6f);
    for (int i = threadIdx.x; i < N; i += 256) {
        float g = gamma ? gamma[i] : 0.f; if (addOne) g += 1.f; else if (!gamma) g = 1.f;
        float b = beta ? beta[i] : 0.f;
        orw[i] = (xr[i] - mu) * inv * g + b;
    }
}

// ---------------- row-wise RMS-norm (*w), optional RoPE, in place ----------------
__global__ __launch_bounds__(256) void rms_rope(
    float* __restrict__ x, const float* __restrict__ w,
    int N, int doRope, int posBase)
{
    const int row = blockIdx.x;
    float* xr = x + (long)row * N;
    __shared__ float red[256];
    float ss = 0.f;
    for (int i = threadIdx.x; i < N; i += 256) { float v = xr[i]; ss += v * v; }
    red[threadIdx.x] = ss; __syncthreads();
    for (int o = 128; o > 0; o >>= 1) { if (threadIdx.x < o) red[threadIdx.x] += red[threadIdx.x + o]; __syncthreads(); }
    const float inv = rsqrtf(red[0] / N + 1e-6f);
    __syncthreads();
    if (!doRope) {
        for (int i = threadIdx.x; i < N; i += 256) xr[i] = xr[i] * inv * w[i];
    } else {
        const float pos = (float)(posBase + row);
        const float lg = logf(10000.f);
        for (int p = threadIdx.x; p < N / 2; p += 256) {   // pairs within each head
            const int h = p >> 6, j = p & 63;
            const int i1 = h * DHEAD + j, i2 = i1 + 64;
            const float x1 = xr[i1] * inv * w[i1];
            const float x2 = xr[i2] * inv * w[i2];
            const float ang = pos * expf(-lg * ((float)j / 64.f));
            const float c = cosf(ang), s = sinf(ang);
            xr[i1] = x1 * c - x2 * s;
            xr[i2] = x1 * s + x2 * c;
        }
    }
}

// ---------------- row-wise scaled softmax, in place ----------------
__global__ __launch_bounds__(256) void softmax_rows(
    float* __restrict__ sc, int N, float scale)
{
    float* r = sc + (long)blockIdx.x * N;
    __shared__ float red[256];
    float mx = -1e30f;
    for (int i = threadIdx.x; i < N; i += 256) mx = fmaxf(mx, r[i] * scale);
    red[threadIdx.x] = mx; __syncthreads();
    for (int o = 128; o > 0; o >>= 1) { if (threadIdx.x < o) red[threadIdx.x] = fmaxf(red[threadIdx.x], red[threadIdx.x + o]); __syncthreads(); }
    const float m = red[0]; __syncthreads();
    float sum = 0.f;
    for (int i = threadIdx.x; i < N; i += 256) { float e = expf(r[i] * scale - m); r[i] = e; sum += e; }
    red[threadIdx.x] = sum; __syncthreads();
    for (int o = 128; o > 0; o >>= 1) { if (threadIdx.x < o) red[threadIdx.x] += red[threadIdx.x + o]; __syncthreads(); }
    const float rs = 1.f / red[0];
    for (int i = threadIdx.x; i < N; i += 256) r[i] *= rs;
}

// ---------------- M=1 vec-mat: out[n] = bias[n] + sum_k act(in[k]) * W[k*N+n] ----------------
__global__ __launch_bounds__(256) void vecmat(
    const float* __restrict__ in, const float* __restrict__ W,
    const float* __restrict__ bias, float* __restrict__ out,
    int K, int N, int siluIn)
{
    const int n = blockIdx.x * 256 + threadIdx.x;
    if (n >= N) return;
    float acc = bias ? bias[n] : 0.f;
    for (int k = 0; k < K; ++k) {
        float v = in[k];
        if (siluIn) v = v / (1.f + expf(-v));
        acc = fmaf(v, W[(long)k * N + n], acc);
    }
    out[n] = acc;
}

// ---------------- tiny helpers ----------------
__global__ void sinemb_kernel(const float* __restrict__ ts, float* __restrict__ out) {
    const int i = threadIdx.x;                // 128 threads
    const float t = ts[0];
    const float fr = expf(-logf(10000.f) * (float)i / 128.f);
    const float a = t * fr;
    out[i] = cosf(a); out[128 + i] = sinf(a);
}

__global__ void add_bcast(const float* __restrict__ a, const float* __restrict__ b,
                          float* __restrict__ out, int n, int bmod) {
    const int e = blockIdx.x * 256 + threadIdx.x;
    if (e >= n) return;
    out[e] = a[e] + b[bmod ? (e % bmod) : e];
}

__global__ void kvcombine(const float* __restrict__ cache, const float* __restrict__ fresh,
                          float* __restrict__ out, int total, int curTokens, int rowLen) {
    const int e = blockIdx.x * 256 + threadIdx.x;
    if (e >= total) return;
    const int kv = e / rowLen, d = e - kv * rowLen;
    out[e] = (kv < curTokens) ? cache[(long)kv * rowLen + d]
                              : fresh[(long)(kv - curTokens) * rowLen + d];
}

__global__ void im2col_patch(const float* __restrict__ x, float* __restrict__ out) {
    const int e = blockIdx.x * 256 + threadIdx.x;   // SEQ*64 = 57344
    if (e >= SEQ * 64) return;
    const int s = e >> 6, c = e & 63;
    const int f = s / 448, r = s % 448, hp = r / 28, wp = r % 28;
    const int cin = c >> 2, ky = (c >> 1) & 1, kx = c & 1;
    out[e] = x[((long)(f * 16 + cin) * HI_H + (hp * 2 + ky)) * WI_W + (wp * 2 + kx)];
}

__global__ void unpatchify(const float* __restrict__ y, float* __restrict__ out) {
    const int e = blockIdx.x * 256 + threadIdx.x;   // NF*COUT*HI*WI = 57344
    if (e >= NF_F * COUT_C * HI_H * WI_W) return;
    const int per = COUT_C * HI_H * WI_W;
    const int f = e / per;
    int rem = e % per;
    const int c = rem / (HI_H * WI_W);
    rem %= HI_H * WI_W;
    const int hh = rem / WI_W, ww = rem % WI_W;
    const int hp = hh >> 1, ky = hh & 1, wp = ww >> 1, kx = ww & 1;
    const int s = f * 448 + hp * 28 + wp;
    out[e] = y[(long)s * 64 + (ky * 2 + kx) * 16 + c];
}

// =================================================================
//                         host orchestration
// =================================================================
extern "C" void kernel_launch(void* const* d_in, const int* in_sizes, int n_in,
                              void* d_out, int out_size, void* d_ws, size_t ws_size,
                              hipStream_t stream)
{
    const float* X        = (const float*)d_in[0];
    const float* TIMESTEP = (const float*)d_in[1];
    const float* PROMPT   = (const float*)d_in[2];
    const float* CACHE_K  = (const float*)d_in[3];
    const float* CACHE_V  = (const float*)d_in[4];
    const float* PATCH_W  = (const float*)d_in[5];
    const float* PATCH_B  = (const float*)d_in[6];
    const float* TEXT_W1  = (const float*)d_in[7];
    const float* TEXT_B1  = (const float*)d_in[8];
    const float* TEXT_W2  = (const float*)d_in[9];
    const float* TEXT_B2  = (const float*)d_in[10];
    const float* TIME_W1  = (const float*)d_in[11];
    const float* TIME_B1  = (const float*)d_in[12];
    const float* TIME_W2  = (const float*)d_in[13];
    const float* TIME_B2  = (const float*)d_in[14];
    const float* TPROJ_W  = (const float*)d_in[15];
    const float* TPROJ_B  = (const float*)d_in[16];
    const float* MOD      = (const float*)d_in[17];
    const float* WQ = (const float*)d_in[18]; const float* WK = (const float*)d_in[19];
    const float* WV = (const float*)d_in[20]; const float* WO = (const float*)d_in[21];
    const float* BQ = (const float*)d_in[22]; const float* BK = (const float*)d_in[23];
    const float* BV = (const float*)d_in[24]; const float* BO = (const float*)d_in[25];
    const float* QN = (const float*)d_in[26]; const float* KN = (const float*)d_in[27];
    const float* CWQ = (const float*)d_in[28]; const float* CWK = (const float*)d_in[29];
    const float* CWV = (const float*)d_in[30]; const float* CWO = (const float*)d_in[31];
    const float* CBQ = (const float*)d_in[32]; const float* CBK = (const float*)d_in[33];
    const float* CBV = (const float*)d_in[34]; const float* CBO = (const float*)d_in[35];
    const float* CQN = (const float*)d_in[36]; const float* CKN = (const float*)d_in[37];
    const float* N3G = (const float*)d_in[38]; const float* N3B = (const float*)d_in[39];
    const float* FW1 = (const float*)d_in[40]; const float* FB1 = (const float*)d_in[41];
    const float* FW2 = (const float*)d_in[42]; const float* FB2 = (const float*)d_in[43];
    const float* HEAD_MOD = (const float*)d_in[44];
    const float* HEAD_W   = (const float*)d_in[45];
    const float* HEAD_B   = (const float*)d_in[46];
    float* OUT = (float*)d_out;

    // ---- workspace carve-up (floats) ----
    float* W = (float*)d_ws;
    size_t off = 0;
    auto take = [&](size_t n) { float* p = W + off; off += n; return p; };
    float* xs    = take((size_t)SEQ * DMODEL);
    float* hn    = take((size_t)SEQ * DMODEL);
    float* qb    = take((size_t)SEQ * DMODEL);
    float* kb    = take((size_t)SEQ * DMODEL);
    float* vb    = take((size_t)SEQ * DMODEL);
    float* obuf  = take((size_t)SEQ * DMODEL);
    float* keys  = take((size_t)KVTOT * DMODEL);
    float* vals  = take((size_t)KVTOT * DMODEL);
    float* ctx   = take((size_t)TLEN * DMODEL);
    float* ctxT  = take((size_t)TLEN * DMODEL);
    float* ctxk  = take((size_t)TLEN * DMODEL);
    float* ctxv  = take((size_t)TLEN * DMODEL);
    float* h1    = take((size_t)SEQ * FFDIM);
    float* scores= take((size_t)NHEAD * SEQ * KVTOT);
    float* semb  = take(256);
    float* temb1 = take(DMODEL);
    float* temb  = take(DMODEL);
    float* e6    = take(6 * DMODEL);
    float* mall  = take((size_t)NLAYER * 6 * DMODEL);
    float* eh    = take(2 * DMODEL);
    float* im2   = take((size_t)SEQ * 64);
    float* ybuf  = take((size_t)SEQ * 64);
    (void)ws_size; (void)n_in; (void)in_sizes; (void)out_size;

    // BN=128 for every GEMM whose N is a multiple of 128; BN=64 for the head.
    auto gemm = [&](const float* A, long lda, long sA,
                    const float* Bp, long ldb, long sB, int trB,
                    float* Cp, long ldc, long sC,
                    const float* bias, const float* scl, const float* res,
                    int M, int N, int K, int act, int batch) {
        if (N % 128 == 0) {
            dim3 g((unsigned)(N / 128), (unsigned)(M / 128), (unsigned)batch);
            gemm_wmma_bf16<128><<<g, 256, 0, stream>>>(A, lda, sA, Bp, ldb, sB, trB,
                                                       Cp, ldc, sC, bias, scl, res, M, N, K, act);
        } else {
            dim3 g((unsigned)(N / 64), (unsigned)(M / 128), (unsigned)batch);
            gemm_wmma_bf16<64><<<g, 256, 0, stream>>>(A, lda, sA, Bp, ldb, sB, trB,
                                                      Cp, ldc, sC, bias, scl, res, M, N, K, act);
        }
    };
    const float SC = 0.08838834764831845f;  // 1/sqrt(128)

    // ---- time embedding chain (M=1) ----
    sinemb_kernel<<<1, 128, 0, stream>>>(TIMESTEP, semb);
    vecmat<<<6, 256, 0, stream>>>(semb,  TIME_W1, TIME_B1, temb1, FDIM,   DMODEL, 0);
    vecmat<<<6, 256, 0, stream>>>(temb1, TIME_W2, TIME_B2, temb,  DMODEL, DMODEL, 1);
    vecmat<<<36, 256, 0, stream>>>(temb, TPROJ_W, TPROJ_B, e6,    DMODEL, 6 * DMODEL, 1);
    add_bcast<<<(NLAYER * 6 * DMODEL + 255) / 256, 256, 0, stream>>>(MOD, e6, mall, NLAYER * 6 * DMODEL, 6 * DMODEL);

    // ---- patch embed: im2col + GEMM(896x1536, K=64, B = patch_w^T) ----
    im2col_patch<<<(SEQ * 64 + 255) / 256, 256, 0, stream>>>(X, im2);
    gemm(im2, 64, 0, PATCH_W, 64, 0, 1, xs, DMODEL, 0, PATCH_B, nullptr, nullptr, SEQ, DMODEL, 64, 0, 1);

    // ---- text context MLP ----
    gemm(PROMPT, TDIM, 0, TEXT_W1, DMODEL, 0, 0, ctxT, DMODEL, 0, TEXT_B1, nullptr, nullptr, TLEN, DMODEL, TDIM, 1, 1);
    gemm(ctxT, DMODEL, 0, TEXT_W2, DMODEL, 0, 0, ctx, DMODEL, 0, TEXT_B2, nullptr, nullptr, TLEN, DMODEL, DMODEL, 0, 1);

    const long DD = (long)DMODEL * DMODEL;
    for (int l = 0; l < NLAYER; ++l) {
        const float* m0 = mall + (size_t)l * 6 * DMODEL;
        const float* m1 = m0 + DMODEL, *m2 = m0 + 2 * DMODEL, *m3 = m0 + 3 * DMODEL,
                   * m4 = m0 + 4 * DMODEL, *m5 = m0 + 5 * DMODEL;

        // ===== self-attention =====
        layernorm_mod<<<SEQ, 256, 0, stream>>>(xs, hn, m1, m0, 1, DMODEL);
        gemm(hn, DMODEL, 0, WQ + l * DD, DMODEL, 0, 0, qb, DMODEL, 0, BQ + l * DMODEL, nullptr, nullptr, SEQ, DMODEL, DMODEL, 0, 1);
        gemm(hn, DMODEL, 0, WK + l * DD, DMODEL, 0, 0, kb, DMODEL, 0, BK + l * DMODEL, nullptr, nullptr, SEQ, DMODEL, DMODEL, 0, 1);
        gemm(hn, DMODEL, 0, WV + l * DD, DMODEL, 0, 0, vb, DMODEL, 0, BV + l * DMODEL, nullptr, nullptr, SEQ, DMODEL, DMODEL, 0, 1);
        rms_rope<<<SEQ, 256, 0, stream>>>(qb, QN + l * DMODEL, DMODEL, 1, CURTOK);
        rms_rope<<<SEQ, 256, 0, stream>>>(kb, KN + l * DMODEL, DMODEL, 1, CURTOK);

        kvcombine<<<(KVTOT * DMODEL + 255) / 256, 256, 0, stream>>>(CACHE_K + (size_t)l * KVCAP * DMODEL, kb, keys, KVTOT * DMODEL, CURTOK, DMODEL);
        kvcombine<<<(KVTOT * DMODEL + 255) / 256, 256, 0, stream>>>(CACHE_V + (size_t)l * KVCAP * DMODEL, vb, vals, KVTOT * DMODEL, CURTOK, DMODEL);

        // scores[h] = q_h (896x128) @ keys_h^T (128x1792)
        gemm(qb, DMODEL, DHEAD, keys, DMODEL, DHEAD, 1, scores, KVTOT, (long)SEQ * KVTOT,
             nullptr, nullptr, nullptr, SEQ, KVTOT, DHEAD, 0, NHEAD);
        softmax_rows<<<NHEAD * SEQ, 256, 0, stream>>>(scores, KVTOT, SC);
        // o_h = P_h (896x1792) @ vals_h (1792x128)
        gemm(scores, KVTOT, (long)SEQ * KVTOT, vals, DMODEL, DHEAD, 0, obuf, DMODEL, DHEAD,
             nullptr, nullptr, nullptr, SEQ, DHEAD, KVTOT, 0, NHEAD);
        // xs += (o @ wo + bo) * m2
        gemm(obuf, DMODEL, 0, WO + l * DD, DMODEL, 0, 0, xs, DMODEL, 0,
             BO + l * DMODEL, m2, xs, SEQ, DMODEL, DMODEL, 0, 1);

        // ===== cross-attention =====
        layernorm_mod<<<SEQ, 256, 0, stream>>>(xs, hn, N3G + l * DMODEL, N3B + l * DMODEL, 0, DMODEL);
        gemm(hn, DMODEL, 0, CWQ + l * DD, DMODEL, 0, 0, qb, DMODEL, 0, CBQ + l * DMODEL, nullptr, nullptr, SEQ, DMODEL, DMODEL, 0, 1);
        rms_rope<<<SEQ, 256, 0, stream>>>(qb, CQN + l * DMODEL, DMODEL, 0, 0);
        gemm(ctx, DMODEL, 0, CWK + l * DD, DMODEL, 0, 0, ctxk, DMODEL, 0, CBK + l * DMODEL, nullptr, nullptr, TLEN, DMODEL, DMODEL, 0, 1);
        rms_rope<<<TLEN, 256, 0, stream>>>(ctxk, CKN + l * DMODEL, DMODEL, 0, 0);
        gemm(ctx, DMODEL, 0, CWV + l * DD, DMODEL, 0, 0, ctxv, DMODEL, 0, CBV + l * DMODEL, nullptr, nullptr, TLEN, DMODEL, DMODEL, 0, 1);

        gemm(qb, DMODEL, DHEAD, ctxk, DMODEL, DHEAD, 1, scores, TLEN, (long)SEQ * TLEN,
             nullptr, nullptr, nullptr, SEQ, TLEN, DHEAD, 0, NHEAD);
        softmax_rows<<<NHEAD * SEQ, 256, 0, stream>>>(scores, TLEN, SC);
        gemm(scores, TLEN, (long)SEQ * TLEN, ctxv, DMODEL, DHEAD, 0, obuf, DMODEL, DHEAD,
             nullptr, nullptr, nullptr, SEQ, DHEAD, TLEN, 0, NHEAD);
        gemm(obuf, DMODEL, 0, CWO + l * DD, DMODEL, 0, 0, xs, DMODEL, 0,
             CBO + l * DMODEL, nullptr, xs, SEQ, DMODEL, DMODEL, 0, 1);

        // ===== FFN =====
        layernorm_mod<<<SEQ, 256, 0, stream>>>(xs, hn, m4, m3, 1, DMODEL);
        gemm(hn, DMODEL, 0, FW1 + (size_t)l * DMODEL * FFDIM, FFDIM, 0, 0, h1, FFDIM, 0,
             FB1 + (size_t)l * FFDIM, nullptr, nullptr, SEQ, FFDIM, DMODEL, 1, 1);
        gemm(h1, FFDIM, 0, FW2 + (size_t)l * FFDIM * DMODEL, DMODEL, 0, 0, xs, DMODEL, 0,
             FB2 + (size_t)l * DMODEL, m5, xs, SEQ, DMODEL, FFDIM, 0, 1);
    }

    // ---- head ----
    add_bcast<<<(2 * DMODEL + 255) / 256, 256, 0, stream>>>(HEAD_MOD, temb, eh, 2 * DMODEL, DMODEL);
    layernorm_mod<<<SEQ, 256, 0, stream>>>(xs, hn, eh + DMODEL, eh, 1, DMODEL);
    gemm(hn, DMODEL, 0, HEAD_W, 64, 0, 0, ybuf, 64, 0, HEAD_B, nullptr, nullptr, SEQ, 64, DMODEL, 0, 1);
    unpatchify<<<(NF_F * COUT_C * HI_H * WI_W + 255) / 256, 256, 0, stream>>>(ybuf, OUT);
}
